// SimCLR_Loss_81716047773781
// MI455X (gfx1250) — matrix operand order repaired and checked by reading, not compiled, not run on hardware
//
#include <hip/hip_runtime.h>

typedef _Float16 v8h  __attribute__((ext_vector_type(8)));
typedef _Float16 v16h __attribute__((ext_vector_type(16)));
typedef float    v8f  __attribute__((ext_vector_type(8)));

#define Dc 256       // feature dim
#define Bc 4096      // batch
#define Nc 8192      // 2*B
// exp(2x - 2) = 2^(x*C - C), C = 2*log2(e)
#define EXP_C 2.8853900817779268f

// ---------------------------------------------------------------------------
// Kernel 1: normalize rows of concat(z_i, z_j) -> f16 (zn16).
// 1 wave per row; each lane handles 8 contiguous floats.
// ---------------------------------------------------------------------------
__global__ void __launch_bounds__(256)
simclr_norm_kernel(const float* __restrict__ zi, const float* __restrict__ zj,
                   _Float16* __restrict__ zn16) {
  const int wave = threadIdx.x >> 5;
  const int lane = threadIdx.x & 31;
  const int row  = blockIdx.x * 8 + wave;

  const float* src = (row < Bc) ? (zi + (size_t)row * Dc)
                                : (zj + (size_t)(row - Bc) * Dc);
  float4 a = *(const float4*)(src + lane * 8);
  float4 b = *(const float4*)(src + lane * 8 + 4);

  float ss = a.x * a.x + a.y * a.y + a.z * a.z + a.w * a.w +
             b.x * b.x + b.y * b.y + b.z * b.z + b.w * b.w;
  for (int off = 16; off > 0; off >>= 1) ss += __shfl_xor(ss, off, 32);

  const float inv = 1.0f / fmaxf(sqrtf(ss), 1e-8f);

  v8h h;
  h[0] = (_Float16)(a.x * inv); h[1] = (_Float16)(a.y * inv);
  h[2] = (_Float16)(a.z * inv); h[3] = (_Float16)(a.w * inv);
  h[4] = (_Float16)(b.x * inv); h[5] = (_Float16)(b.y * inv);
  h[6] = (_Float16)(b.z * inv); h[7] = (_Float16)(b.w * inv);
  *(v8h*)(zn16 + (size_t)row * Dc + lane * 8) = h;
}

// ---------------------------------------------------------------------------
// B-tile load: 16 cols x 256 K of f16 in WMMA B-layout (lane = N, elements
// 0..7 <-> K = half*8 + 0..7, 8..15 <-> K = 16 + half*8 + 0..7) -> 8 v16h, two b128 loads each.
// ---------------------------------------------------------------------------
__device__ __forceinline__ void load_btile(const _Float16* __restrict__ zn16,
                                           int c0, int nn, int half,
                                           v16h (&Bv)[8]) {
  const _Float16* brow = zn16 + (size_t)(c0 + nn) * Dc + half * 8;
#pragma unroll
  for (int s = 0; s < 8; ++s) {
    v8h lo = *(const v8h*)(brow + s * 32);
    v8h hi = *(const v8h*)(brow + s * 32 + 16);
    Bv[s] = __builtin_shufflevector(lo, hi, 0, 1, 2, 3, 4, 5, 6, 7,
                                            8, 9, 10, 11, 12, 13, 14, 15);
  }
}

// 8-deep wmma chain over K=256, then branch-free exp epilogue.
__device__ __forceinline__ void tile_mma(const v16h (&A)[8], const v16h (&Bv)[8],
                                         v8f& rowsum) {
  v8f acc = {0.f, 0.f, 0.f, 0.f, 0.f, 0.f, 0.f, 0.f};
#pragma unroll
  for (int s = 0; s < 8; ++s)
    acc = __builtin_amdgcn_wmma_f32_16x16x32_f16(
        false, A[s], false, Bv[s], (short)0, acc, false, false);
#pragma unroll
  for (int p = 0; p < 8; ++p)
    rowsum[p] += exp2f(fmaf(acc[p], EXP_C, -EXP_C));   // exp(sim - 2)
}

// ---------------------------------------------------------------------------
// Kernel 2: sim = zn @ zn^T / T, streaming sum(exp(sim-2)) per row
// (diagonal INCLUDED; subtracted exactly in the finalize kernel).
// Grid: 64 row-blocks (128 rows) x 8 column chunks (1024 cols).
// Workgroup: 8 waves, wave w owns rows r0=rb*128+16w (A resident in VGPRs);
// all waves sweep the same column tiles (redundant B loads hit WGP cache).
// Double-buffered B prefetch hides load latency behind the wmma chain.
// Deterministic shfl reduction; no atomics.
// ---------------------------------------------------------------------------
__global__ void __launch_bounds__(256)
simclr_sim_kernel(const _Float16* __restrict__ zn16,
                  float* __restrict__ srow_part) {
  const int rb   = blockIdx.x >> 3;
  const int cb   = blockIdx.x & 7;
  const int wave = threadIdx.x >> 5;
  const int lane = threadIdx.x & 31;
  const int half = lane >> 4;
  const int nn   = lane & 15;
  const int r0   = rb * 128 + wave * 16;
  const int c_base = cb * 1024;

  // --- preload A: 16 rows x 256 K in WMMA A-layout (lane = M) ---
  v16h Areg[8];
  const _Float16* arow = zn16 + (size_t)(r0 + nn) * Dc;
#pragma unroll
  for (int s = 0; s < 8; ++s) {
    v8h lo = *(const v8h*)(arow + s * 32 + half * 8);       // K = half*8+0..7
    v8h hi = *(const v8h*)(arow + s * 32 + half * 8 + 16);  // K = 16+half*8+0..7
    Areg[s] = __builtin_shufflevector(lo, hi, 0, 1, 2, 3, 4, 5, 6, 7,
                                              8, 9, 10, 11, 12, 13, 14, 15);
  }

  v8f rowsum = {0.f, 0.f, 0.f, 0.f, 0.f, 0.f, 0.f, 0.f};

  v16h B0[8], B1[8];
  load_btile(zn16, c_base, nn, half, B0);
  for (int t = 0; t < 64; t += 2) {
    load_btile(zn16, c_base + (t + 1) * 16, nn, half, B1);   // prefetch t+1
    tile_mma(Areg, B0, rowsum);                              // compute t
    const int tn = (t + 2 < 64) ? (t + 2) : 63;              // clamped prefetch
    load_btile(zn16, c_base + tn * 16, nn, half, B0);
    tile_mma(Areg, B1, rowsum);                              // compute t+1
  }

  // Deterministic cross-lane reduction: sum over the 16 lanes sharing `half`
  // (xor masks 1,2,4,8 stay within the half-group).
#pragma unroll
  for (int off = 1; off < 16; off <<= 1)
#pragma unroll
    for (int p = 0; p < 8; ++p)
      rowsum[p] += __shfl_xor(rowsum[p], off, 32);

  // lanes 0 and 16 hold full chunk-partials for rows r0+8*half+0..7
  if (nn == 0) {
    float* dst = srow_part + (size_t)cb * Nc + r0 + 8 * half;
    float4 lo4 = {rowsum[0], rowsum[1], rowsum[2], rowsum[3]};
    float4 hi4 = {rowsum[4], rowsum[5], rowsum[6], rowsum[7]};
    *(float4*)(dst)     = lo4;
    *(float4*)(dst + 4) = hi4;
  }
}

// ---------------------------------------------------------------------------
// Kernel 3: per-row finalize. 1 wave per row.
//  self = ||zn16_i||^2 (f16 products, f32 acc)  -> subtract exp(2*self-2)
//  dot  = zn16_i . zn16_{i^B}                   -> positive = 2*dot
//  term = log( sum_part - exp(2*self-2) ) + 2 - 2*dot
// ---------------------------------------------------------------------------
__global__ void __launch_bounds__(256)
simclr_finalize_kernel(const _Float16* __restrict__ zn16,
                       const float* __restrict__ srow_part,
                       float* __restrict__ terms) {
  const int wave = threadIdx.x >> 5;
  const int lane = threadIdx.x & 31;
  const int row  = blockIdx.x * 8 + wave;
  const int prt  = row ^ Bc;   // (row + B) mod N for power-of-two N

  v8h a = *(const v8h*)(zn16 + (size_t)row * Dc + lane * 8);
  v8h b = *(const v8h*)(zn16 + (size_t)prt * Dc + lane * 8);
  float dot = 0.f, self = 0.f;
#pragma unroll
  for (int k = 0; k < 8; ++k) {
    float av = (float)a[k], bv = (float)b[k];
    dot  = fmaf(av, bv, dot);
    self = fmaf(av, av, self);
  }
  for (int off = 16; off > 0; off >>= 1) {
    dot  += __shfl_xor(dot, off, 32);
    self += __shfl_xor(self, off, 32);
  }

  if (lane == 0) {
    float s = 0.f;
#pragma unroll
    for (int cb = 0; cb < 8; ++cb) s += srow_part[(size_t)cb * Nc + row];
    s -= exp2f(fmaf(self, EXP_C, -EXP_C));      // drop self-similarity term
    terms[row] = logf(s) + 2.0f - 2.0f * dot;   // lse - positive
  }
}

// ---------------------------------------------------------------------------
// Kernel 4: loss = mean(terms), deterministic LDS tree reduction.
// ---------------------------------------------------------------------------
__global__ void __launch_bounds__(256)
simclr_reduce_kernel(const float* __restrict__ terms, float* __restrict__ out) {
  __shared__ float red[256];
  float acc = 0.0f;
  for (int i = threadIdx.x; i < Nc; i += 256) acc += terms[i];
  red[threadIdx.x] = acc;
  __syncthreads();
  for (int s = 128; s > 0; s >>= 1) {
    if (threadIdx.x < s) red[threadIdx.x] += red[threadIdx.x + s];
    __syncthreads();
  }
  if (threadIdx.x == 0) out[0] = red[0] / (float)Nc;
}

// ---------------------------------------------------------------------------
extern "C" void kernel_launch(void* const* d_in, const int* in_sizes, int n_in,
                              void* d_out, int out_size, void* d_ws, size_t ws_size,
                              hipStream_t stream) {
  (void)in_sizes; (void)n_in; (void)out_size; (void)ws_size;

  const float* zi = (const float*)d_in[0];
  const float* zj = (const float*)d_in[1];
  float* out = (float*)d_out;

  // workspace: zn16 (4 MB) | srow_part (8*N floats, 256 KB) | terms (32 KB)
  char* w = (char*)d_ws;
  _Float16* zn16   = (_Float16*)w;
  float* srow_part = (float*)(w + (size_t)Nc * Dc * sizeof(_Float16));
  float* terms     = srow_part + (size_t)8 * Nc;

  simclr_norm_kernel<<<dim3(Nc / 8), dim3(256), 0, stream>>>(zi, zj, zn16);
  simclr_sim_kernel<<<dim3(64 * 8), dim3(256), 0, stream>>>(zn16, srow_part);
  simclr_finalize_kernel<<<dim3(Nc / 8), dim3(256), 0, stream>>>(zn16, srow_part, terms);
  simclr_reduce_kernel<<<dim3(1), dim3(256), 0, stream>>>(terms, out);
}
